// GCN_Node_44272522887302
// MI455X (gfx1250) — matrix operand order repaired and checked by reading, not compiled, not run on hardware
//
#include <hip/hip_runtime.h>

typedef __attribute__((ext_vector_type(16))) _Float16 v16h;
typedef __attribute__((ext_vector_type(8)))  float    v8f;
typedef __attribute__((ext_vector_type(4)))  unsigned u32x4;
typedef __attribute__((ext_vector_type(8)))  int      i32x8;
typedef __attribute__((ext_vector_type(4)))  int      i32x4;

#if __has_builtin(__builtin_amdgcn_tensor_load_to_lds) && __has_builtin(__builtin_amdgcn_s_wait_tensorcnt)
#define USE_TDM 1
#else
#define USE_TDM 0
#endif

// ---------------------------------------------------------------------------
// small utility kernels
// ---------------------------------------------------------------------------
__global__ void k_fill_f32(float* __restrict__ p, float v, long n) {
    long i = blockIdx.x * (long)blockDim.x + threadIdx.x;
    if (i < n) p[i] = v;
}

__global__ void k_deg_count(const long long* __restrict__ col,
                            float* __restrict__ deg, long E) {
    long e = blockIdx.x * (long)blockDim.x + threadIdx.x;
    if (e < E) unsafeAtomicAdd(&deg[(int)col[e]], 1.0f);
}

__global__ void k_rsqrt_inplace(float* __restrict__ p, long n) {
    long i = blockIdx.x * (long)blockDim.x + threadIdx.x;
    if (i < n) p[i] = rsqrtf(p[i]);
}

__global__ void k_f32_to_f16(const float* __restrict__ src,
                             _Float16* __restrict__ dst, long n) {
    long i = blockIdx.x * (long)blockDim.x + threadIdx.x;
    if (i < n) dst[i] = (_Float16)src[i];
}

// W[K,N] (f32, row-major) -> Wt[N,K] (f16) so both GEMM operands are K-contiguous.
__global__ void k_w_to_f16t(const float* __restrict__ W, _Float16* __restrict__ Wt,
                            int K, int N) {
    long i = blockIdx.x * (long)blockDim.x + threadIdx.x;
    if (i >= (long)K * N) return;
    int k = (int)(i / N), n = (int)(i % N);
    Wt[(long)n * K + k] = (_Float16)W[i];
}

// ---------------------------------------------------------------------------
// TDM helper: issue a 2D tile load (tile1 rows x tile0 elems, f16) into LDS.
// D# built per CDNA5 ISA §8: group0 {count=1, lds_addr, global_addr, type=2},
// group1 {data_size=2B, tensor_dim0/1 (remaining extents, OOB reads 0),
//         tile_dim0/1, tensor_dim0_stride}. Groups 2/3 (+extra group) zero:
// tile_dim2/3/4 = 0 -> dims unused; 2D tile mode.
// ---------------------------------------------------------------------------
#if USE_TDM
__device__ __forceinline__ unsigned lds_addr_of(const void* p) {
    return (unsigned)(unsigned long long)(__attribute__((address_space(3))) const void*)p;
}

__device__ __forceinline__ void tdm_load_tile(const _Float16* gsrc, unsigned lds_byte,
                                              int rem0, int rem1,
                                              int tile0, int tile1, long stride0) {
    unsigned long long ga = (unsigned long long)(const void*)gsrc;
    u32x4 g0;
    g0[0] = 1u;                                             // count=1, user mode
    g0[1] = lds_byte;                                       // lds_addr
    g0[2] = (unsigned)(ga & 0xFFFFFFFFu);                   // global_addr[31:0]
    g0[3] = (unsigned)((ga >> 32) & 0x1FFFFFFu) | (2u << 30); // addr[56:32] | type=2
    i32x8 g1;
    g1[0] = 1 << 16;                                        // data_size=1 -> 2 bytes
    g1[1] = (rem0 & 0xFFFF) << 16;                          // tensor_dim0[15:0]
    g1[2] = ((rem0 >> 16) & 0xFFFF) | ((rem1 & 0xFFFF) << 16); // dim0 hi | dim1 lo
    g1[3] = ((rem1 >> 16) & 0xFFFF) | (tile0 << 16);        // dim1 hi | tile_dim0
    g1[4] = tile1;                                          // tile_dim1 (tile_dim2=0)
    g1[5] = (int)(stride0 & 0xFFFFFFFF);                    // dim0 stride lo
    g1[6] = (int)((stride0 >> 32) & 0xFFFF);                // dim0 stride hi
    g1[7] = 0;
    i32x4 z4 = {0, 0, 0, 0};
    i32x8 z8 = {0, 0, 0, 0, 0, 0, 0, 0};
    __builtin_amdgcn_tensor_load_to_lds(g0, g1, z4, z4, z8, 0);
}
#endif

// ---------------------------------------------------------------------------
// WMMA GEMM: C[M,N] = A[M,K] (f16) * Bt[N,K]^T (f16), f32 accumulate.
// Block = 256 threads = 8 waves; block tile 128(M) x 32(N); each wave computes
// a 16x32 slab (two 16x16 WMMA tiles reusing the A fragment).
// Tiles double-buffered in LDS, fed by TDM (tensor_load_to_lds) when available.
// K % 32 == 0, N % 32 == 0.
// ---------------------------------------------------------------------------
__launch_bounds__(256)
__global__ void k_gemm_wmma(const _Float16* __restrict__ A,
                            const _Float16* __restrict__ Bt,
                            float* __restrict__ C,
                            int M, int N, int K) {
    __shared__ _Float16 ldsA[2][128 * 32];   // [buf][m][k]  (8 KB each)
    __shared__ _Float16 ldsB[2][32 * 32];    // [buf][n][k]  (2 KB each)

    const int tid    = threadIdx.x;
    const int lane   = tid & 31;
    const int wv     = tid >> 5;
    const int blockM = blockIdx.y * 128;
    const int blockN = blockIdx.x * 32;

    const int khalf = lane >> 4;
    const int nloc  = lane & 15;
    const int mloc  = (wv << 4) + nloc;

    const int nIter = K >> 5;

    v8f acc0 = {};
    v8f acc1 = {};

#if USE_TDM
    if (tid < 32) {
        tdm_load_tile(A + (long)blockM * K, lds_addr_of(&ldsA[0][0]),
                      K, M - blockM, 32, 128, K);
        tdm_load_tile(Bt + (long)blockN * K, lds_addr_of(&ldsB[0][0]),
                      K, N - blockN, 32, 32, K);
    }
#endif

    for (int it = 0; it < nIter; ++it) {
        const int buf = it & 1;
        const int k0  = it << 5;

#if USE_TDM
        if (tid < 32) __builtin_amdgcn_s_wait_tensorcnt(0);
        __syncthreads();                     // tiles[buf] ready; prior reads of buf^1 done
        if (tid < 32 && it + 1 < nIter) {
            const int k1 = k0 + 32;
            tdm_load_tile(A + (long)blockM * K + k1, lds_addr_of(&ldsA[buf ^ 1][0]),
                          K - k1, M - blockM, 32, 128, K);
            tdm_load_tile(Bt + (long)blockN * K + k1, lds_addr_of(&ldsB[buf ^ 1][0]),
                          K - k1, N - blockN, 32, 32, K);
        }
#else
        // ---- synchronous fallback staging (clamped rows, b128 transfers) ----
        {
            const uint4* __restrict__ Ag = (const uint4*)A;
            uint4* la = (uint4*)&ldsA[buf][0];
            #pragma unroll
            for (int i = 0; i < 2; ++i) {
                int d = tid + i * 256;       // 0..511 -> 128 rows x 4 xb128
                int r = d >> 2, q = d & 3;
                int gm = blockM + r; if (gm >= M) gm = M - 1;  // clamp: row only feeds its own (guarded) C row
                la[d] = Ag[((long)gm * K + k0) / 8 + q];
            }
            if (tid < 128) {
                const uint4* __restrict__ Bg = (const uint4*)Bt;
                uint4* lb = (uint4*)&ldsB[buf][0];
                int r = tid >> 2, q = tid & 3;
                lb[tid] = Bg[((long)(blockN + r) * K + k0) / 8 + q];
            }
            if (it + 1 < nIter) {
                int pr = blockM + (tid >> 1); if (pr >= M) pr = M - 1;
                __builtin_prefetch(A + (long)pr * K + k0 + 32, 0, 1);
            }
        }
        __syncthreads();
#endif

        // ---- fragments: contiguous k order, same for A and B ----
        // A 16x32 f16: VGPR v: K = khalf*16+2v(+s)
        // B 32x16 f16: lanes 0-15 hold K=0..15, lanes 16-31 K=16..31; VGPR v: K=khalf*16+2v(+s)
        const unsigned* la_u = (const unsigned*)&ldsA[buf][0];
        const unsigned* lb_u = (const unsigned*)&ldsB[buf][0];
        union { v16h v; unsigned u[8]; } af, bf0, bf1;
        #pragma unroll
        for (int v = 0; v < 8; ++v) {
            int ka = khalf * 16 + 2 * v;
            af.u[v]  = la_u[mloc * 16 + (ka >> 1)];
            bf0.u[v] = lb_u[(nloc + 0) * 16 + khalf * 8 + v];
            bf1.u[v] = lb_u[(nloc + 16) * 16 + khalf * 8 + v];
        }

        acc0 = __builtin_amdgcn_wmma_f32_16x16x32_f16(false, af.v, false, bf0.v,
                                                      (short)0, acc0, false, false);
        acc1 = __builtin_amdgcn_wmma_f32_16x16x32_f16(false, af.v, false, bf1.v,
                                                      (short)0, acc1, false, false);
#if !USE_TDM
        __syncthreads();
#endif
    }

    // ---- store: C/D layout: VGPR r -> M = (lane>=16 ? 8:0)+r, N = lane&15 ----
    union { v8f v; float f[8]; } c0, c1;
    c0.v = acc0; c1.v = acc1;
    const int mbase = blockM + (wv << 4) + khalf * 8;
    const int gn0 = blockN + nloc;
    #pragma unroll
    for (int r = 0; r < 8; ++r) {
        int gm = mbase + r;
        if (gm < M) {
            C[(long)gm * N + gn0]      = c0.f[r];
            C[(long)gm * N + gn0 + 16] = c1.f[r];
        }
    }
}

// ---------------------------------------------------------------------------
// Edge-parallel normalized scatter-add, float4 gathers + 4 atomics/thread:
//   agg[col[e]*C + c] += dinv[row[e]]*dinv[col[e]] * xlin[row[e]*C + c]
// C = 1<<logC (>=4). Atomics stay L2-resident (working set << 192MB L2).
// ---------------------------------------------------------------------------
__global__ void k_scatter(const float4* __restrict__ xlin4,
                          const long long* __restrict__ row,
                          const long long* __restrict__ col,
                          const float* __restrict__ dinv,
                          float* __restrict__ agg,
                          long E, int logC) {
    const int logQ = logC - 2;               // float4 groups per node
    long t = blockIdx.x * (long)blockDim.x + threadIdx.x;
    long total = E << logQ;
    if (t >= total) return;
    long e = t >> logQ;
    int  q = (int)(t & ((1L << logQ) - 1));
    int  r  = (int)row[e];
    int  cl = (int)col[e];
    float norm = dinv[r] * dinv[cl];
    float4 v = xlin4[((long)r << logQ) + q];
    float* dst = agg + ((long)cl << logC) + (q << 2);
    unsafeAtomicAdd(dst + 0, norm * v.x);
    unsafeAtomicAdd(dst + 1, norm * v.y);
    unsafeAtomicAdd(dst + 2, norm * v.z);
    unsafeAtomicAdd(dst + 3, norm * v.w);
}

// ---------------------------------------------------------------------------
// Epilogue: v = agg + dinv^2 * xlin + bias ; optional sigmoid;
// writes f16 activation for the next layer and/or f32 final output.
// ---------------------------------------------------------------------------
__global__ void k_epilogue(const float* __restrict__ agg,
                           const float* __restrict__ xlin,
                           const float* __restrict__ dinv,
                           const float* __restrict__ bias,
                           _Float16* __restrict__ h16_out,
                           float* __restrict__ f32_out,
                           long n, int logC, int do_sigmoid) {
    long i = blockIdx.x * (long)blockDim.x + threadIdx.x;
    if (i >= n) return;
    long m = i >> logC;
    int  c = (int)(i & ((1L << logC) - 1));
    float di = dinv[m];
    float v = agg[i] + di * di * xlin[i] + bias[c];
    if (do_sigmoid) v = 1.0f / (1.0f + __expf(-v));
    if (h16_out) h16_out[i] = (_Float16)v;
    if (f32_out) f32_out[i] = v;
}

// ---------------------------------------------------------------------------
// host launcher
// ---------------------------------------------------------------------------
static inline long cdivl(long a, long b) { return (a + b - 1) / b; }

extern "C" void kernel_launch(void* const* d_in, const int* in_sizes, int n_in,
                              void* d_out, int out_size, void* d_ws, size_t ws_size,
                              hipStream_t stream) {
    const float*     x  = (const float*)d_in[0];
    // d_in[1] = edge_attr (unused by forward)
    const long long* ei = (const long long*)d_in[2];
    const float* W1 = (const float*)d_in[3];
    const float* b1 = (const float*)d_in[4];
    const float* W2 = (const float*)d_in[5];
    const float* b2 = (const float*)d_in[6];
    const float* W3 = (const float*)d_in[7];
    const float* b3 = (const float*)d_in[8];

    const int  IN_CH = 128, HID = 256, OUT_CH = 64;
    const long Nn = in_sizes[0] / IN_CH;     // 50000
    const long E  = in_sizes[2] / 2;         // 800000
    const long long* row = ei;
    const long long* col = ei + E;

    // ---- workspace carve-up ----
    char* ws = (char*)d_ws;
    size_t off = 0;
    auto carve = [&](size_t bytes) -> void* {
        void* p = ws + off;
        off += (bytes + 255) & ~(size_t)255;
        return p;
    };
    float*    dinv = (float*)   carve(Nn * sizeof(float));
    _Float16* h16  = (_Float16*)carve(Nn * HID * sizeof(_Float16));
    float*    xlin = (float*)   carve(Nn * HID * sizeof(float));
    float*    agg  = (float*)   carve(Nn * HID * sizeof(float));
    _Float16* w16a = (_Float16*)carve((size_t)IN_CH * HID * sizeof(_Float16));  // [HID][IN_CH]
    _Float16* w16b = (_Float16*)carve((size_t)HID * HID * sizeof(_Float16));    // [HID][HID]
    _Float16* w16c = (_Float16*)carve((size_t)HID * OUT_CH * sizeof(_Float16)); // [OUT][HID]
    (void)ws_size;

    const int TB = 256;

    // ---- degrees: deg = 1 (self loop) + count(col) ; dinv = rsqrt(deg) ----
    k_fill_f32<<<cdivl(Nn, TB), TB, 0, stream>>>(dinv, 1.0f, Nn);
    k_deg_count<<<cdivl(E, TB), TB, 0, stream>>>(col, dinv, E);
    k_rsqrt_inplace<<<cdivl(Nn, TB), TB, 0, stream>>>(dinv, Nn);

    // ---- f16 conversions (weights transposed to [N][K]) ----
    k_f32_to_f16<<<cdivl(Nn * IN_CH, TB), TB, 0, stream>>>(x, h16, Nn * IN_CH);
    k_w_to_f16t<<<cdivl((long)IN_CH * HID, TB), TB, 0, stream>>>(W1, w16a, IN_CH, HID);
    k_w_to_f16t<<<cdivl((long)HID * HID, TB), TB, 0, stream>>>(W2, w16b, HID, HID);
    k_w_to_f16t<<<cdivl((long)HID * OUT_CH, TB), TB, 0, stream>>>(W3, w16c, HID, OUT_CH);

    const long gridMy = cdivl(Nn, 128);

    // ================= layer 1: 128 -> 256, sigmoid =================
    k_gemm_wmma<<<dim3(HID / 32, gridMy), TB, 0, stream>>>(h16, w16a, xlin, (int)Nn, HID, IN_CH);
    k_fill_f32<<<cdivl(Nn * HID, TB), TB, 0, stream>>>(agg, 0.0f, Nn * HID);
    k_scatter<<<cdivl(E << 6, TB), TB, 0, stream>>>((const float4*)xlin, row, col, dinv, agg, E, 8);
    k_epilogue<<<cdivl(Nn * HID, TB), TB, 0, stream>>>(agg, xlin, dinv, b1,
                                                       h16, (float*)nullptr, Nn * HID, 8, 1);

    // ================= layer 2: 256 -> 256, sigmoid =================
    k_gemm_wmma<<<dim3(HID / 32, gridMy), TB, 0, stream>>>(h16, w16b, xlin, (int)Nn, HID, HID);
    k_fill_f32<<<cdivl(Nn * HID, TB), TB, 0, stream>>>(agg, 0.0f, Nn * HID);
    k_scatter<<<cdivl(E << 6, TB), TB, 0, stream>>>((const float4*)xlin, row, col, dinv, agg, E, 8);
    k_epilogue<<<cdivl(Nn * HID, TB), TB, 0, stream>>>(agg, xlin, dinv, b2,
                                                       h16, (float*)nullptr, Nn * HID, 8, 1);

    // ================= layer 3: 256 -> 64, linear out =================
    k_gemm_wmma<<<dim3(OUT_CH / 32, gridMy), TB, 0, stream>>>(h16, w16c, xlin, (int)Nn, OUT_CH, HID);
    k_fill_f32<<<cdivl(Nn * OUT_CH, TB), TB, 0, stream>>>(agg, 0.0f, Nn * OUT_CH);
    k_scatter<<<cdivl(E << 4, TB), TB, 0, stream>>>((const float4*)xlin, row, col, dinv, agg, E, 6);
    k_epilogue<<<cdivl(Nn * OUT_CH, TB), TB, 0, stream>>>(agg, xlin, dinv, b3,
                                                          (_Float16*)nullptr, (float*)d_out,
                                                          Nn * OUT_CH, 6, 0);
}
